// series_decomp_ema_momentum_66331474920167
// MI455X (gfx1250) — matrix operand order repaired and checked by reading, not compiled, not run on hardware
//
#include <hip/hip_runtime.h>
#include <hip/hip_bf16.h>
#include <math.h>

// ---------------- problem constants ----------------
#define BB 8
#define LL 4096
#define DD 512
#define NA 3
#define NC 32            // time chunks for scans
#define CHUNK 128        // LL / NC
#define KSTEPS 16        // DD / 32  (K per WMMA = 32)
#define NTILES 32        // DD / 16  (N per WMMA = 16)
#define LN_EPS 1e-5f

typedef __bf16 bf16_t;
typedef bf16_t v16bf __attribute__((ext_vector_type(16)));
typedef bf16_t v8bf  __attribute__((ext_vector_type(8)));
typedef float  v8f   __attribute__((ext_vector_type(8)));

union V16 { v16bf v; struct { v8bf lo, hi; } h; };

__device__ __forceinline__ float sigmoidf_(float x) { return 1.0f / (1.0f + expf(-x)); }

// ---------------- kernel 0: f32 -> bf16 weight conversion ----------------
__global__ void cvt_bf16_kernel(const float* __restrict__ src, bf16_t* __restrict__ dst, int n) {
    int i = blockIdx.x * blockDim.x + threadIdx.x;
    if (i < n) dst[i] = (bf16_t)src[i];
}

// ---------------- EMA scan, 3-phase chunked (affine recurrence) ----------------
// thread mapping for phase 1/3: tid -> (b, c, d), d fastest (coalesced on x)
__global__ void ema_phase1_kernel(const float* __restrict__ x,
                                  const float* __restrict__ logits,
                                  float* __restrict__ E) {
    int tid = blockIdx.x * blockDim.x + threadIdx.x;    // BB*NC*DD
    int d = tid % DD;
    int c = (tid / DD) % NC;
    int b = tid / (DD * NC);
    float al[NA];
    #pragma unroll
    for (int a = 0; a < NA; ++a) al[a] = sigmoidf_(logits[a]);

    const float* xb = x + (size_t)b * LL * DD + d;
    int t0 = c * CHUNK;
    float xprev = (t0 == 0) ? 0.0f : xb[(size_t)(t0 - 1) * DD];
    float m[NA] = {0.0f, 0.0f, 0.0f};
    for (int i = 0; i < CHUNK; ++i) {
        int t = t0 + i;
        float xt = xb[(size_t)t * DD];
        float v = (t == 0) ? (xb[DD] - xt) : (xt - xprev);
        xprev = xt;
        if (t == 0) {
            #pragma unroll
            for (int a = 0; a < NA; ++a) m[a] = v;          // m0 = v0 exactly
        } else {
            #pragma unroll
            for (int a = 0; a < NA; ++a) m[a] = al[a] * m[a] + (1.0f - al[a]) * v;
        }
    }
    #pragma unroll
    for (int a = 0; a < NA; ++a)
        E[(((size_t)b * NC + c) * NA + a) * DD + d] = m[a];
}

__global__ void ema_phase2_kernel(const float* __restrict__ logits,
                                  const float* __restrict__ E,
                                  float* __restrict__ S) {
    int tid = blockIdx.x * blockDim.x + threadIdx.x;    // BB*DD
    int d = tid % DD;
    int b = tid / DD;
    for (int a = 0; a < NA; ++a) {
        float al = sigmoidf_(logits[a]);
        float F = powf(al, (float)CHUNK);               // chunk decay factor
        float s = 0.0f;
        for (int c = 0; c < NC; ++c) {
            size_t idx = (((size_t)b * NC + c) * NA + a) * DD + d;
            S[idx] = s;                                 // carry-in for chunk c
            float Ec = E[idx];
            s = (c == 0) ? Ec : (F * s + Ec);           // chunk 0 ignores carry (t==0 override)
        }
    }
}

__global__ void ema_phase3_kernel(const float* __restrict__ x,
                                  const float* __restrict__ logits,
                                  const float* __restrict__ S,
                                  bf16_t* __restrict__ fused) {
    int tid = blockIdx.x * blockDim.x + threadIdx.x;    // BB*NC*DD
    int d = tid % DD;
    int c = (tid / DD) % NC;
    int b = tid / (DD * NC);
    float al[NA], w[NA];
    float wsum = 0.0f;
    #pragma unroll
    for (int a = 0; a < NA; ++a) { al[a] = sigmoidf_(logits[a]); w[a] = expf(al[a]); wsum += w[a]; }
    #pragma unroll
    for (int a = 0; a < NA; ++a) w[a] /= wsum;          // softmax over alphas

    float m[NA];
    #pragma unroll
    for (int a = 0; a < NA; ++a) m[a] = S[(((size_t)b * NC + c) * NA + a) * DD + d];

    const float* xb = x + (size_t)b * LL * DD + d;
    bf16_t* fb = fused + (size_t)b * LL * DD + d;
    int t0 = c * CHUNK;
    float xprev = (t0 == 0) ? 0.0f : xb[(size_t)(t0 - 1) * DD];
    for (int i = 0; i < CHUNK; ++i) {
        int t = t0 + i;
        float xt = xb[(size_t)t * DD];
        float v = (t == 0) ? (xb[DD] - xt) : (xt - xprev);
        xprev = xt;
        if (t == 0) {
            #pragma unroll
            for (int a = 0; a < NA; ++a) m[a] = v;
        } else {
            #pragma unroll
            for (int a = 0; a < NA; ++a) m[a] = al[a] * m[a] + (1.0f - al[a]) * v;
        }
        float f = w[0] * m[0] + w[1] * m[1] + w[2] * m[2];
        fb[(size_t)t * DD] = (bf16_t)f;
    }
}

// ---------------- WMMA MLP: GEMM1 -> LN -> GELU -> GEMM2 ----------------
// 4 waves/block, each wave owns 16 tokens. A tile (16x512 bf16) resident in
// VGPRs as 16 v16bf frags in contiguous k order (k = 16*half + s), same as B. B frags are
// per-lane contiguous 32B loads from bf16 weights (row n = output channel).
__global__ __launch_bounds__(128) void mlp_wmma_kernel(
    const bf16_t* __restrict__ fused,
    const bf16_t* __restrict__ w1, const float* __restrict__ b1,
    const float* __restrict__ lnw, const float* __restrict__ lnb,
    const bf16_t* __restrict__ w2, const float* __restrict__ b2,
    float* __restrict__ trend_raw)
{
    extern __shared__ char smem[];                       // 48KB per wave
    const int lane = threadIdx.x & 31;
    const int wave = threadIdx.x >> 5;
    float*  hf = (float*)(smem + wave * (48 * 1024));            // 16x512 f32
    bf16_t* hb = (bf16_t*)(smem + wave * (48 * 1024) + 32 * 1024); // 16x512 bf16

    const int tok0 = (blockIdx.x * 4 + wave) * 16;
    const int mrow = lane & 15;          // A row / C column within tile
    const int half = lane >> 4;          // lane half selects K sub-range / C row half

    // ---- stage A fragments for GEMM1 (resident in registers) ----
    v16bf afrag[KSTEPS];
    {
        const bf16_t* arow = fused + (size_t)(tok0 + mrow) * DD;
        #pragma unroll
        for (int ks = 0; ks < KSTEPS; ++ks) {
            V16 t;
            t.h.lo = *(const v8bf*)(arow + ks * 32 + half * 16);       // K = half*16 ..
            t.h.hi = *(const v8bf*)(arow + ks * 32 + half * 16 + 8);   // K = half*16+8 ..
            afrag[ks] = t.v;
        }
    }

    // ---- GEMM1: h = fused @ w1^T + b1  -> LDS (f32) ----
    for (int nt = 0; nt < NTILES; ++nt) {
        const int n = nt * 16 + mrow;
        float bias = b1[n];
        v8f acc;
        #pragma unroll
        for (int r = 0; r < 8; ++r) acc[r] = bias;
        const bf16_t* brow = w1 + (size_t)n * DD + half * 16;    // B[k][n] = w1[n][k]
        #pragma unroll
        for (int ks = 0; ks < KSTEPS; ++ks) {
            v16bf bfrag = *(const v16bf*)(brow + ks * 32);
            acc = __builtin_amdgcn_wmma_f32_16x16x32_bf16(
                      false, afrag[ks], false, bfrag, (short)0, acc, false, false);
        }
        #pragma unroll
        for (int r = 0; r < 8; ++r)                    // C layout: lane n, rows r+8*half
            hf[(r + 8 * half) * DD + n] = acc[r];
    }
    __syncthreads();

    // ---- LayerNorm + exact GELU per row, emit bf16 ----
    for (int m = 0; m < 16; ++m) {
        float s = 0.0f, s2 = 0.0f;
        #pragma unroll
        for (int j = 0; j < 16; ++j) {
            float v = hf[m * DD + lane + j * 32];
            s += v; s2 += v * v;
        }
        #pragma unroll
        for (int off = 16; off > 0; off >>= 1) {
            s  += __shfl_xor(s,  off, 32);
            s2 += __shfl_xor(s2, off, 32);
        }
        float mu = s * (1.0f / DD);
        float var = s2 * (1.0f / DD) - mu * mu;
        float rs = rsqrtf(var + LN_EPS);
        #pragma unroll
        for (int j = 0; j < 16; ++j) {
            int cidx = lane + j * 32;
            float v = (hf[m * DD + cidx] - mu) * rs * lnw[cidx] + lnb[cidx];
            v = 0.5f * v * (1.0f + erff(v * 0.70710678118654752f));
            hb[m * DD + cidx] = (bf16_t)v;
        }
    }
    __syncthreads();

    // ---- stage A fragments for GEMM2 from LDS (ds_load_b128) ----
    {
        const bf16_t* arow = hb + (size_t)mrow * DD;
        #pragma unroll
        for (int ks = 0; ks < KSTEPS; ++ks) {
            V16 t;
            t.h.lo = *(const v8bf*)(arow + ks * 32 + half * 16);
            t.h.hi = *(const v8bf*)(arow + ks * 32 + half * 16 + 8);
            afrag[ks] = t.v;
        }
    }

    // ---- GEMM2: trend_raw = h @ w2^T + b2 -> global f32 ----
    for (int nt = 0; nt < NTILES; ++nt) {
        const int n = nt * 16 + mrow;
        float bias = b2[n];
        v8f acc;
        #pragma unroll
        for (int r = 0; r < 8; ++r) acc[r] = bias;
        const bf16_t* brow = w2 + (size_t)n * DD + half * 16;
        #pragma unroll
        for (int ks = 0; ks < KSTEPS; ++ks) {
            v16bf bfrag = *(const v16bf*)(brow + ks * 32);
            acc = __builtin_amdgcn_wmma_f32_16x16x32_bf16(
                      false, afrag[ks], false, bfrag, (short)0, acc, false, false);
        }
        #pragma unroll
        for (int r = 0; r < 8; ++r)
            trend_raw[(size_t)(tok0 + r + 8 * half) * DD + n] = acc[r];
    }
}

// ---------------- cumulative sum over L, 3-phase, in place ----------------
__global__ void cum_phase1_kernel(const float* __restrict__ tr, float* __restrict__ cs) {
    int tid = blockIdx.x * blockDim.x + threadIdx.x;    // BB*NC*DD
    int d = tid % DD;
    int c = (tid / DD) % NC;
    int b = tid / (DD * NC);
    const float* p = tr + ((size_t)b * LL + c * CHUNK) * DD + d;
    float s = 0.0f;
    for (int i = 0; i < CHUNK; ++i) s += p[(size_t)i * DD];
    cs[((size_t)b * NC + c) * DD + d] = s;
}

__global__ void cum_phase2_kernel(float* __restrict__ cs) {
    int tid = blockIdx.x * blockDim.x + threadIdx.x;    // BB*DD
    int d = tid % DD;
    int b = tid / DD;
    float run = 0.0f;
    for (int c = 0; c < NC; ++c) {
        size_t idx = ((size_t)b * NC + c) * DD + d;
        float t = cs[idx];
        cs[idx] = run;                                   // exclusive prefix
        run += t;
    }
}

__global__ void cum_phase3_kernel(float* __restrict__ tr, const float* __restrict__ cs) {
    int tid = blockIdx.x * blockDim.x + threadIdx.x;    // BB*NC*DD
    int d = tid % DD;
    int c = (tid / DD) % NC;
    int b = tid / (DD * NC);
    float run = cs[((size_t)b * NC + c) * DD + d];
    float* p = tr + ((size_t)b * LL + c * CHUNK) * DD + d;
    for (int i = 0; i < CHUNK; ++i) {
        run += p[(size_t)i * DD];
        p[(size_t)i * DD] = run;                         // in-place cumsum
    }
}

// ---------------- depthwise conv (k=5, zero pad 2) + anchor + outputs ----------------
__global__ void conv_seasonal_kernel(const float* __restrict__ x,
                                     const float* __restrict__ convw,
                                     const float* __restrict__ tc,
                                     float* __restrict__ out) {
    size_t idx = (size_t)blockIdx.x * blockDim.x + threadIdx.x;   // BB*LL*DD
    int d = (int)(idx % DD);
    size_t r = idx / DD;
    int l = (int)(r % LL);
    int b = (int)(r / LL);
    size_t base = (size_t)b * LL * DD + d;
    float anchor = x[base] - tc[base];                   // tc[b,0,d] == trend_raw[b,0,d]
    float acc = 0.0f;
    #pragma unroll
    for (int j = 0; j < 5; ++j) {
        int lj = l + j - 2;
        if (lj >= 0 && lj < LL)
            acc += convw[d * 5 + j] * (tc[base + (size_t)lj * DD] + anchor);
    }
    float xv = x[idx];
    out[idx] = xv - acc;                                 // seasonal
    out[(size_t)BB * LL * DD + idx] = acc;               // trend
}

// ---------------- launcher ----------------
extern "C" void kernel_launch(void* const* d_in, const int* in_sizes, int n_in,
                              void* d_out, int out_size, void* d_ws, size_t ws_size,
                              hipStream_t stream) {
    const float* x      = (const float*)d_in[0];
    const float* logits = (const float*)d_in[1];
    const float* w1     = (const float*)d_in[2];
    const float* b1     = (const float*)d_in[3];
    const float* lnw    = (const float*)d_in[4];
    const float* lnb    = (const float*)d_in[5];
    const float* w2     = (const float*)d_in[6];
    const float* b2     = (const float*)d_in[7];
    const float* convw  = (const float*)d_in[8];
    float* out = (float*)d_out;

    // workspace layout
    char* ws = (char*)d_ws;
    const size_t n_elem = (size_t)BB * LL * DD;          // 16M
    bf16_t* fused = (bf16_t*)ws;                         // 32 MB
    float*  trend = (float*)(ws + n_elem * 2);           // 64 MB
    bf16_t* w1b   = (bf16_t*)(ws + n_elem * 2 + n_elem * 4);
    bf16_t* w2b   = w1b + DD * DD;
    float*  E     = (float*)((char*)(w2b + DD * DD));    // B*NC*NA*D
    float*  S     = E + (size_t)BB * NC * NA * DD;
    float*  CS    = S + (size_t)BB * NC * NA * DD;       // B*NC*D

    // 0) weights -> bf16
    cvt_bf16_kernel<<<(DD * DD + 255) / 256, 256, 0, stream>>>(w1, w1b, DD * DD);
    cvt_bf16_kernel<<<(DD * DD + 255) / 256, 256, 0, stream>>>(w2, w2b, DD * DD);

    // 1) EMA momentum scan -> fused (bf16)
    const int scan_threads = BB * NC * DD;               // 131072
    ema_phase1_kernel<<<scan_threads / 256, 256, 0, stream>>>(x, logits, E);
    ema_phase2_kernel<<<(BB * DD) / 256, 256, 0, stream>>>(logits, E, S);
    ema_phase3_kernel<<<scan_threads / 256, 256, 0, stream>>>(x, logits, S, fused);

    // 2) WMMA MLP -> trend_raw (f32)
    const int tokens = BB * LL;                          // 32768
    mlp_wmma_kernel<<<tokens / 64, 128, 4 * 48 * 1024, stream>>>(
        fused, w1b, b1, lnw, lnb, w2b, b2, trend);

    // 3) cumsum over L (in place on trend)
    cum_phase1_kernel<<<scan_threads / 256, 256, 0, stream>>>(trend, CS);
    cum_phase2_kernel<<<(BB * DD) / 256, 256, 0, stream>>>(CS);
    cum_phase3_kernel<<<scan_threads / 256, 256, 0, stream>>>(trend, CS);

    // 4) depthwise conv + anchor + seasonal/trend outputs
    conv_seasonal_kernel<<<(unsigned)(n_elem / 256), 256, 0, stream>>>(x, convw, trend, out);
}
